// LocalAttention1_42588895707684
// MI455X (gfx1250) — matrix operand order repaired and checked by reading, compiled, not run on hardware
//
#include <hip/hip_runtime.h>
#include <hip/hip_bf16.h>

// ---------------------------------------------------------------------------
// Local (annulus-masked) attention for MI455X / gfx1250.
// All matmuls on v_wmma_f32_16x16x32_bf16 (fp32 accumulate).
// b=2, n=2048, dim=512, heads=8, dim_head=64, window=256, max_range=1024.
//
// Data-movement strategy:
//  * fp32 operands are converted to bf16 ONCE (Xb, Wqkv^T, Wout^T).
//  * All A-operands (Xb, Q, attn-out, P-tile in LDS) are stored "A-swizzled":
//    each 32-wide K chunk is kept in plain order (lane half h reads K 16h..16h+15),
//    so a lane's full 16-element WMMA A fragment is ONE contiguous 32B load.
//  * All B-operands (Wqkv^T, Wout^T, K row-major, V transposed) are contiguous
//    in the contraction dim -> ONE contiguous 32B load per fragment.
// ---------------------------------------------------------------------------

typedef __attribute__((ext_vector_type(16))) __bf16 v16bf;
typedef __attribute__((ext_vector_type(8)))  float  v8f;

static constexpr int BATCH = 2;
static constexpr int NSEQ  = 2048;
static constexpr int DIM   = 512;
static constexpr int HEADS = 8;
static constexpr int DHEAD = 64;
static constexpr int INNER = HEADS * DHEAD;   // 512
static constexpr int QKVN  = 3 * INNER;       // 1536
static constexpr int MROWS = BATCH * NSEQ;    // 4096
static constexpr int WIN   = 256;
static constexpr int MAXR  = 1024;

// fp32 -> bf16 RNE (bit ops only)
__device__ __forceinline__ __bf16 f2bf(float f) {
  unsigned u = __float_as_uint(f);
  unsigned r = u + 0x7FFFu + ((u >> 16) & 1u);
  unsigned short h = (unsigned short)(r >> 16);
  __bf16 b;
  __builtin_memcpy(&b, &h, sizeof(b));
  return b;
}

// A layout within a 32-element K chunk: plain order (lane half h reads K 16h..16h+15).
// stored_pos(K) = K
__device__ __forceinline__ int aswz(int c) {
  return c;
}

__device__ __forceinline__ v8f wmma_bf16(v16bf a, v16bf b, v8f c) {
  return __builtin_amdgcn_wmma_f32_16x16x32_bf16(false, a, false, b, (short)0, c,
                                                 false, false);
}

// row reductions across a 16-lane half (xor masks 1,2,4,8 stay inside halves)
__device__ __forceinline__ float rowmax16(float v) {
#pragma unroll
  for (int m = 1; m <= 8; m <<= 1) v = fmaxf(v, __shfl_xor(v, m, 32));
  return v;
}
__device__ __forceinline__ float rowsum16(float v) {
#pragma unroll
  for (int m = 1; m <= 8; m <<= 1) v += __shfl_xor(v, m, 32);
  return v;
}

// ---------------------------------------------------------------------------
// Pre-conversion kernels (run once, bandwidth-trivial: ~18 MB total)
// ---------------------------------------------------------------------------
// X [4096,512] fp32 -> A-swizzled bf16
__global__ __launch_bounds__(256) void cvt_x_kernel(const float* __restrict__ X,
                                                    __bf16* __restrict__ Xb) {
  int idx = blockIdx.x * 256 + threadIdx.x;          // one element each
  int row = idx >> 9, col = idx & 511;
  Xb[((size_t)row << 9) + (col & ~31) + aswz(col & 31)] = f2bf(X[idx]);
}

// W [K,N] fp32 -> Wt [N,K] bf16 (plain order along K: B-fragment friendly)
__global__ __launch_bounds__(256) void cvt_wt_kernel(const float* __restrict__ W,
                                                     __bf16* __restrict__ Wt,
                                                     int K, int N) {
  int idx = blockIdx.x * 256 + threadIdx.x;          // idx = n*K + k
  int n = idx / K, k = idx % K;
  Wt[idx] = f2bf(W[(size_t)k * N + n]);
}

// ---------------------------------------------------------------------------
// Kernel 1: QKV = Xb @ Wqkv^T, emitted as bf16 Q (A-swizzled, x scale),
// K (row-major [n,64]) and V transposed ([64,n]).
// Block: 256 thr (8 waves); block tile 128(M) x 64(N); wave tile 16 x 64.
// ---------------------------------------------------------------------------
__global__ __launch_bounds__(256) void qkv_proj_kernel(const __bf16* __restrict__ Xb,
                                                       const __bf16* __restrict__ Wt,
                                                       __bf16* __restrict__ Qb,
                                                       __bf16* __restrict__ Kb,
                                                       __bf16* __restrict__ Vtb) {
  const int lane = threadIdx.x & 31;
  const int wv   = threadIdx.x >> 5;
  const int half = lane >> 4;
  const int ln   = lane & 15;

  const int bm = blockIdx.x / (QKVN / 64);
  const int bn = blockIdx.x % (QKVN / 64);
  const int m0 = bm * 128 + wv * 16;
  const int c0 = bn * 64;

  const v8f zero = {0.f, 0.f, 0.f, 0.f, 0.f, 0.f, 0.f, 0.f};
  v8f acc[4] = {zero, zero, zero, zero};

  const __bf16* arow = Xb + (size_t)(m0 + ln) * DIM;
  for (int k = 0; k < DIM; k += 32) {
    v16bf a = *(const v16bf*)(arow + k + half * 16);               // A-swizzled
#pragma unroll
    for (int t = 0; t < 4; ++t) {
      v16bf b = *(const v16bf*)(Wt + (size_t)(c0 + t * 16 + ln) * DIM + k + half * 16);
      acc[t] = wmma_bf16(a, b, acc[t]);
    }
  }

#pragma unroll
  for (int t = 0; t < 4; ++t) {
    const int c     = c0 + t * 16 + ln;
    const int which = c / INNER;            // 0=q 1=k 2=v (uniform per col-tile)
    const int h     = (c % INNER) / DHEAD;
    const int d     = c % DHEAD;
#pragma unroll
    for (int v = 0; v < 8; ++v) {
      const int r  = m0 + v + half * 8;     // global row in [0, 4096)
      const int b  = r / NSEQ;
      const int nn = r % NSEQ;
      const int bh = b * HEADS + h;
      const float f = acc[t][v];
      if (which == 0)
        Qb[((size_t)bh * NSEQ + nn) * DHEAD + (d & ~31) + aswz(d & 31)] = f2bf(f * 0.125f);
      else if (which == 1)
        Kb[((size_t)bh * NSEQ + nn) * DHEAD + d] = f2bf(f);
      else
        Vtb[((size_t)bh * DHEAD + d) * NSEQ + nn] = f2bf(f);
    }
  }
}

// ---------------------------------------------------------------------------
// Kernel 2: flash-attention with annulus mask (attend iff d<=256 || d>1024).
// Block: 256 thr (8 waves); one (b,h), 128 queries per block; 16 q / wave.
// Key loop over 32-key blocks; wave-uniform skip of fully-masked tiles.
// ---------------------------------------------------------------------------
__global__ __launch_bounds__(256) void attn_kernel(const __bf16* __restrict__ Q,
                                                   const __bf16* __restrict__ K,
                                                   const __bf16* __restrict__ Vt,
                                                   __bf16* __restrict__ Aout) {
  // per-wave 16x32 P tile, A-swizzled rows, no cross-wave sharing (no barriers)
  __shared__ __attribute__((aligned(32))) __bf16 p_lds[8][16][32];

  const int lane = threadIdx.x & 31;
  const int wv   = threadIdx.x >> 5;
  const int half = lane >> 4;
  const int ln   = lane & 15;

  const int qt = blockIdx.x & 15;           // 16 query tiles of 128 per (b,h)
  const int bh = blockIdx.x >> 4;           // 0..15
  const int q0 = qt * 128 + wv * 16;

  const __bf16* Qp = Q  + (size_t)bh * NSEQ * DHEAD;
  const __bf16* Kp = K  + (size_t)bh * NSEQ * DHEAD;
  const __bf16* Vp = Vt + (size_t)bh * DHEAD * NSEQ;

  // Q fragments (A-swizzled storage -> single 32B load each); scale folded in.
  v16bf qa0, qa1;
  {
    const __bf16* rowp = Qp + (size_t)(q0 + ln) * DHEAD;
    qa0 = *(const v16bf*)(rowp + half * 16);
    qa1 = *(const v16bf*)(rowp + 32 + half * 16);
  }

  const v8f zero = {0.f, 0.f, 0.f, 0.f, 0.f, 0.f, 0.f, 0.f};
  v8f o[4] = {zero, zero, zero, zero};
  float m_i[8], l_i[8];
#pragma unroll
  for (int v = 0; v < 8; ++v) { m_i[v] = -1e30f; l_i[v] = 0.f; }

  // LDS store positions for the P tile (plain order of key%32)
  const int pos0 = ln;        // key K = ln
  const int pos1 = 16 + ln;   // key K = 16 + ln

  for (int j0 = 0; j0 < NSEQ; j0 += 32) {
    // wave-uniform tile classification against the annulus mask
    const int dmin = (j0 > q0 + 15) ? (j0 - (q0 + 15))
                                    : ((q0 > j0 + 31) ? (q0 - (j0 + 31)) : 0);
    int da = q0 - (j0 + 31); if (da < 0) da = -da;
    int db = (q0 + 15) - j0; if (db < 0) db = -db;
    const int dmax = (da > db) ? da : db;
    if (dmin > WIN && dmax <= MAXR) continue;            // fully masked tile
    const bool need_mask = !((dmax <= WIN) || (dmin > MAXR));

    // S = Q K^T for this 16q x 32key tile (two 16x16 C tiles)
    v8f s[2] = {zero, zero};
#pragma unroll
    for (int jt = 0; jt < 2; ++jt) {
      const __bf16* kp = Kp + (size_t)(j0 + jt * 16 + ln) * DHEAD + half * 16;
      v16bf kb0 = *(const v16bf*)kp;           // d = half*16 + e
      v16bf kb1 = *(const v16bf*)(kp + 32);    // d = 32 + half*16 + e
      s[jt] = wmma_bf16(qa0, kb0, s[jt]);
      s[jt] = wmma_bf16(qa1, kb1, s[jt]);
    }

    if (need_mask) {
#pragma unroll
      for (int v = 0; v < 8; ++v) {
        const int i = q0 + v + half * 8;
        int d0 = i - (j0 + ln);        if (d0 < 0) d0 = -d0;
        int d1 = i - (j0 + 16 + ln);   if (d1 < 0) d1 = -d1;
        const bool a0 = (d0 <= WIN) || (d0 > MAXR);
        const bool a1 = (d1 <= WIN) || (d1 > MAXR);
        s[0][v] = a0 ? s[0][v] : -1e30f;
        s[1][v] = a1 ? s[1][v] : -1e30f;
      }
    }

    // online softmax (per row = per accumulator slot v within a lane-half)
    float alpha[8];
#pragma unroll
    for (int v = 0; v < 8; ++v) {
      float mx = rowmax16(fmaxf(s[0][v], s[1][v]));
      const float m_new = fmaxf(m_i[v], mx);
      alpha[v] = __expf(m_i[v] - m_new);
      m_i[v] = m_new;
      const float p0 = __expf(s[0][v] - m_new);
      const float p1 = __expf(s[1][v] - m_new);
      s[0][v] = p0; s[1][v] = p1;
      l_i[v] = l_i[v] * alpha[v] + rowsum16(p0 + p1);
    }
#pragma unroll
    for (int t = 0; t < 4; ++t)
#pragma unroll
      for (int v = 0; v < 8; ++v) o[t][v] *= alpha[v];

    // reshape P: C-layout -> A-fragment via wave-private LDS tile (A-swizzled)
#pragma unroll
    for (int v = 0; v < 8; ++v) {
      p_lds[wv][v + half * 8][pos0] = f2bf(s[0][v]);
      p_lds[wv][v + half * 8][pos1] = f2bf(s[1][v]);
    }
    asm volatile("s_wait_dscnt 0" ::: "memory");
    v16bf pa = *(const v16bf*)&p_lds[wv][ln][half * 16];  // single 32B ds read

    // O += P @ V  (V transposed: contiguous over keys)
#pragma unroll
    for (int t = 0; t < 4; ++t) {
      const __bf16* vp = Vp + (size_t)(t * 16 + ln) * NSEQ + j0 + half * 16;
      v16bf vb = *(const v16bf*)vp;
      o[t] = wmma_bf16(pa, vb, o[t]);
    }
  }

  // epilogue: normalize, emit [b, n, h*64 + d] bf16 (A-swizzled for kernel 3)
  const int bb = bh >> 3, hh = bh & 7;
#pragma unroll
  for (int t = 0; t < 4; ++t) {
#pragma unroll
    for (int v = 0; v < 8; ++v) {
      const int i = q0 + v + half * 8;
      const int d = t * 16 + ln;
      const float val = o[t][v] / l_i[v];
      Aout[((size_t)(bb * NSEQ + i)) * INNER + hh * DHEAD + (d & ~31) + aswz(d & 31)]
          = f2bf(val);
    }
  }
}

// ---------------------------------------------------------------------------
// Kernel 3: Out = Attn @ Wout^T + bias  (fp32 result)
// ---------------------------------------------------------------------------
__global__ __launch_bounds__(256) void out_proj_kernel(const __bf16* __restrict__ Ain,
                                                       const __bf16* __restrict__ Wot,
                                                       const float* __restrict__ bias,
                                                       float* __restrict__ Out) {
  const int lane = threadIdx.x & 31;
  const int wv   = threadIdx.x >> 5;
  const int half = lane >> 4;
  const int ln   = lane & 15;

  const int bm = blockIdx.x / (DIM / 64);
  const int bn = blockIdx.x % (DIM / 64);
  const int m0 = bm * 128 + wv * 16;
  const int c0 = bn * 64;

  const v8f zero = {0.f, 0.f, 0.f, 0.f, 0.f, 0.f, 0.f, 0.f};
  v8f acc[4] = {zero, zero, zero, zero};

  const __bf16* arow = Ain + (size_t)(m0 + ln) * INNER;
  for (int k = 0; k < INNER; k += 32) {
    v16bf a = *(const v16bf*)(arow + k + half * 16);               // A-swizzled
#pragma unroll
    for (int t = 0; t < 4; ++t) {
      v16bf b = *(const v16bf*)(Wot + (size_t)(c0 + t * 16 + ln) * INNER + k + half * 16);
      acc[t] = wmma_bf16(a, b, acc[t]);
    }
  }

#pragma unroll
  for (int t = 0; t < 4; ++t) {
    const int c = c0 + t * 16 + ln;
    const float bo = bias[c];
#pragma unroll
    for (int v = 0; v < 8; ++v) {
      const int r = m0 + v + half * 8;
      Out[(size_t)r * DIM + c] = acc[t][v] + bo;
    }
  }
}

// ---------------------------------------------------------------------------
extern "C" void kernel_launch(void* const* d_in, const int* in_sizes, int n_in,
                              void* d_out, int out_size, void* d_ws, size_t ws_size,
                              hipStream_t stream) {
  (void)in_sizes; (void)n_in; (void)out_size; (void)ws_size;
  const float* x     = (const float*)d_in[0];
  const float* w_qkv = (const float*)d_in[1];
  const float* w_out = (const float*)d_in[2];
  const float* b_out = (const float*)d_in[3];
  float* out = (float*)d_out;

  // workspace layout (bf16 elements)
  const size_t TEN = (size_t)BATCH * HEADS * NSEQ * DHEAD;  // 2,097,152 (4 MB)
  __bf16* Xb    = (__bf16*)d_ws;                 // 4096*512
  __bf16* Wqt   = Xb   + (size_t)MROWS * DIM;    // 1536*512
  __bf16* Wot   = Wqt  + (size_t)QKVN * DIM;     // 512*512
  __bf16* Qb    = Wot  + (size_t)DIM * INNER;
  __bf16* Kb    = Qb + TEN;
  __bf16* Vtb   = Kb + TEN;
  __bf16* Ab    = Vtb + TEN;

  cvt_x_kernel <<<(MROWS * DIM) / 256, 256, 0, stream>>>(x, Xb);
  cvt_wt_kernel<<<(QKVN * DIM) / 256, 256, 0, stream>>>(w_qkv, Wqt, DIM, QKVN);
  cvt_wt_kernel<<<(DIM * INNER) / 256, 256, 0, stream>>>(w_out, Wot, INNER, DIM);

  qkv_proj_kernel<<<(MROWS / 128) * (QKVN / 64), 256, 0, stream>>>(Xb, Wqt, Qb, Kb, Vtb);
  attn_kernel<<<BATCH * HEADS * (NSEQ / 128), 256, 0, stream>>>(Qb, Kb, Vtb, Ab);
  out_proj_kernel<<<(MROWS / 128) * (DIM / 64), 256, 0, stream>>>(Ab, w_out ? Wot : Wot, b_out, out);
}
